// Spectralawaregate_60859686584644
// MI455X (gfx1250) — compile-verified, matrix operand order checked by reading, not run on hardware
//
#include <hip/hip_runtime.h>

// ---------------- types ----------------
typedef __attribute__((ext_vector_type(16))) __bf16 v16bf;
typedef __attribute__((ext_vector_type(8)))  float  v8f;
typedef __attribute__((ext_vector_type(4)))  unsigned int u32x4;
typedef __attribute__((ext_vector_type(4)))  int i32x4;
typedef __attribute__((ext_vector_type(8)))  int i32x8;

union Bf16Frag { v16bf v; unsigned int u[8]; };

__device__ __forceinline__ unsigned short f32_to_bf16(float f) {
  unsigned int u = __float_as_uint(f);
  unsigned int r = u + 0x7FFFu + ((u >> 16) & 1u);   // round-to-nearest-even
  return (unsigned short)(r >> 16);
}

// ---------------- constants ----------------
#define BH 2
#define HH 128
#define WW 128
#define CC 256
#define CS 64
#define NPIX (BH * HH * WW)          // 32768
#define BN_EPS 1e-3f

// ============================================================
// Kernel 1: gate = relu(BN1( (avg3x3 + max3x3)(fmapS) @ w1 ))
// 4 pixels per 256-thread block; 64 threads (one per Cs channel) per pixel.
// ============================================================
__global__ __launch_bounds__(256) void gate_kernel(
    const float* __restrict__ fmapS, const float* __restrict__ w1,
    const float* __restrict__ g1, const float* __restrict__ b1,
    const float* __restrict__ m1, const float* __restrict__ v1,
    float* __restrict__ gate) {
  __shared__ float sarr[4][CS];
  const int tid = threadIdx.x;
  const int pl = tid >> 6;         // pixel within block (0..3)
  const int c  = tid & 63;         // Cs channel
  const int P  = blockIdx.x * 4 + pl;
  const int b  = P >> 14;
  const int rem = P & 16383;
  const int oh = rem >> 7, ow = rem & 127;

  float sum = 0.f, mx = -3.402823466e+38f;
  int cnt = 0;
  #pragma unroll
  for (int di = -1; di <= 1; ++di) {
    int y = oh + di;
    if ((unsigned)y >= (unsigned)HH) continue;
    #pragma unroll
    for (int dj = -1; dj <= 1; ++dj) {
      int x = ow + dj;
      if ((unsigned)x >= (unsigned)WW) continue;
      float s = fmapS[(((b * HH + y) * WW) + x) * CS + c];
      sum += s;
      mx = fmaxf(mx, s);
      ++cnt;
    }
  }
  sarr[pl][c] = sum / (float)cnt + mx;   // TF SAME avg (count excl. pad) + max
  __syncthreads();

  if (tid < 36) {                  // 4 pixels * 9 gate channels
    const int p2 = tid / 9, f = tid % 9;
    float acc = 0.f;
    #pragma unroll 8
    for (int cc = 0; cc < CS; ++cc) acc += sarr[p2][cc] * w1[cc * 9 + f];
    const float sc = g1[f] * rsqrtf(v1[f] + BN_EPS);
    float x = (acc - m1[f]) * sc + b1[f];
    x = fmaxf(x, 0.f);
    gate[(blockIdx.x * 4 + p2) * 9 + f] = x;
  }
}

// ============================================================
// Kernel 2: repack w2 (HWIO f32) -> bf16 blocks w2p[(t*8+kb)][n][32]
//   t = dy*3+dx (tap), kb = K-block (K=Cin), inner 32 = contiguous K.
//   Each 16 KB block is contiguous => one TDM descriptor per (tap, kblock).
// ============================================================
__global__ __launch_bounds__(256) void w2prep_kernel(
    const float* __restrict__ w2, unsigned short* __restrict__ w2p) {
  const int tid = blockIdx.x * 256 + threadIdx.x;   // < 589824
  const int kk = tid & 31;
  const int n  = (tid >> 5) & 255;
  const int kb = (tid >> 13) & 7;
  const int t  = tid >> 16;
  const int k  = kb * 32 + kk;
  const float f = w2[(t * CC + k) * CC + n];        // w2[dy][dx][k][n]
  w2p[tid] = f32_to_bf16(f);
}

// ============================================================
// TDM: issue a contiguous 16 KB (2048 x 8B) tensor load into LDS.
// ============================================================
__device__ __forceinline__ void tdm_load_16k(const void* gptr, unsigned lds_off) {
  unsigned long long ga = (unsigned long long)(uintptr_t)gptr;
  u32x4 g0;
  g0.x = 1u;                                   // count=1 (valid), user mode
  g0.y = lds_off;                              // LDS byte address
  g0.z = (unsigned)ga;                         // global_addr[31:0]
  g0.w = (unsigned)(ga >> 32) | (2u << 30);    // global_addr[56:32] | type=2
  i32x8 g1;
  g1[0] = (3 << 16);                           // data_size=3 (8 bytes/elem)
  g1[1] = (2048 << 16);                        // tensor_dim0 = 2048
  g1[2] = (1 << 16);                           // tensor_dim1 = 1
  g1[3] = (2048 << 16);                        // tile_dim0 = 2048
  g1[4] = 0;                                   // tile_dim1/2 unused
  g1[5] = 2048;                                // tensor_dim0_stride
  g1[6] = 0;
  g1[7] = 0;
  i32x4 z4 = {0, 0, 0, 0};
#if defined(__clang_major__) && (__clang_major__ >= 23)
  i32x8 z8 = {0, 0, 0, 0, 0, 0, 0, 0};
  __builtin_amdgcn_tensor_load_to_lds(g0, g1, z4, z4, z8, 0);
#else
  __builtin_amdgcn_tensor_load_to_lds(g0, g1, z4, z4, 0);
#endif
}

// ============================================================
// Kernel 3: 9-tap gated GEMM with bf16 WMMA, TDM double-buffered weights.
// Grid: 512 blocks x 256 thr. Block = 64 pixels x 256 Cout.
// Wave (8 total): wm(0..1) x wn(0..3) -> 32 pixels x 64 Cout.
// Per wave: 2 M sub-blocks x 4 N tiles = 8 accumulators; each B fragment
// is reused by both M sub-blocks (8 WMMA per 12 ds_load_b128).
// ============================================================
__global__ __launch_bounds__(256) void conv_kernel(
    const float* __restrict__ fmapD, const float* __restrict__ gate,
    const unsigned short* __restrict__ w2p,
    const float* __restrict__ g2, const float* __restrict__ b2,
    const float* __restrict__ m2, const float* __restrict__ v2,
    float* __restrict__ out) {
  __shared__ unsigned short Atile[64 * 256];       // 32 KB gated bf16 A
  __shared__ unsigned short Btile[2][256 * 32];    // 2 x 16 KB weight K-blocks

  const int tid  = threadIdx.x;
  const int lane = tid & 31;
  const int wave = tid >> 5;
  const int wm   = wave >> 2;            // M half (0..1): rows wm*32 .. +31
  const int wn   = wave & 3;             // N quarter (0..3)
  const int nBase = wn * 64;
  const int pixBase = blockIdx.x * 64;
  const bool lead = (wave == 0);

  v8f acc[2][4];
  #pragma unroll
  for (int mb = 0; mb < 2; ++mb)
    #pragma unroll
    for (int j = 0; j < 4; ++j)
      acc[mb][j] = (v8f){0.f,0.f,0.f,0.f,0.f,0.f,0.f,0.f};

  // Per-lane K offsets inside a 32-wide K block (wave32 bf16 A/B layout):
  // lanes 0-15 own K pairs {0,1..6,7, 16,17..22,23}; lanes 16-31 shift by 8.
  int koffs[8];
  {
    const int kb2 = (lane >> 4) << 3;
    #pragma unroll
    for (int v = 0; v < 8; ++v)
      koffs[v] = ((v < 4) ? (2 * v) : (8 + 2 * v)) + kb2;
  }

  // Prologue: TDM-load first weight block into buffer 0.
  if (lead) tdm_load_16k(w2p, (unsigned)(uintptr_t)&Btile[0][0]);

  for (int t = 0; t < 9; ++t) {
    // ---- build gated bf16 A tile: 64 rows x 256 K ----
    {
      const int r  = tid >> 2;           // row (pixel) 0..63
      const int c0 = (tid & 3) << 6;     // 64 channels per thread
      const int P  = pixBase + r;
      const int b  = P >> 14, rem = P & 16383;
      const int oh = rem >> 7, ow = rem & 127;
      const int dy = t / 3, dx = t - 3 * dy;
      const int rem2 = 384 * dy + 3 * ow + dx;   // TF reshape index algebra
      const int wp = rem2 / 9;
      const int kk = rem2 - 9 * wp;
      const int kh = kk / 3, kw = kk - 3 * kh;
      const float gv = gate[((b * HH + oh) * WW + wp) * 9 + kk];
      const int sy = oh + kh - 1, sx = wp + kw - 1;
      unsigned int* dst = (unsigned int*)&Atile[r * 256 + c0];
      if ((unsigned)sy < (unsigned)HH && (unsigned)sx < (unsigned)WW) {
        const float4* src = (const float4*)(fmapD +
            ((size_t)((b * HH + sy) * WW + sx)) * CC + c0);
        #pragma unroll
        for (int q = 0; q < 16; ++q) {
          float4 fv = src[q];
          unsigned short h0 = f32_to_bf16(fv.x * gv);
          unsigned short h1 = f32_to_bf16(fv.y * gv);
          unsigned short h2 = f32_to_bf16(fv.z * gv);
          unsigned short h3 = f32_to_bf16(fv.w * gv);
          dst[q * 2 + 0] = (unsigned)h0 | ((unsigned)h1 << 16);
          dst[q * 2 + 1] = (unsigned)h2 | ((unsigned)h3 << 16);
        }
      } else {
        #pragma unroll
        for (int q = 0; q < 32; ++q) dst[q] = 0u;  // zero padding rows
      }
      // Prefetch next tap's source row for this pixel (global_prefetch_b8).
      if (t < 8) {
        const int t1 = t + 1;
        const int dy1 = t1 / 3, dx1 = t1 - 3 * dy1;
        const int rem3 = 384 * dy1 + 3 * ow + dx1;
        const int wp1 = rem3 / 9;
        const int kk1 = rem3 - 9 * wp1;
        const int kh1 = kk1 / 3, kw1 = kk1 - 3 * kh1;
        const int sy1 = oh + kh1 - 1, sx1 = wp1 + kw1 - 1;
        if ((unsigned)sy1 < (unsigned)HH && (unsigned)sx1 < (unsigned)WW)
          __builtin_prefetch(fmapD +
              ((size_t)((b * HH + sy1) * WW + sx1)) * CC + c0, 0, 1);
      }
    }
    __syncthreads();   // A ready (prev tap's reads finished at loop-tail barrier)

    for (int kb = 0; kb < 8; ++kb) {
      const int i = t * 8 + kb;
      if (lead) {
        if (i + 1 < 72) {
          tdm_load_16k(w2p + (size_t)(i + 1) * 8192,
                       (unsigned)(uintptr_t)&Btile[(i + 1) & 1][0]);
          __builtin_amdgcn_s_wait_tensorcnt((short)1);  // block i complete (in-order)
        } else {
          __builtin_amdgcn_s_wait_tensorcnt((short)0);
        }
      }
      __syncthreads();                 // B[i&1] visible to all waves

      const unsigned short* Bb = &Btile[i & 1][0];

      Bf16Frag af[2];
      #pragma unroll
      for (int mb = 0; mb < 2; ++mb) {
        const int arow = (wm * 32 + mb * 16 + (lane & 15)) * 256 + kb * 32;
        #pragma unroll
        for (int v = 0; v < 8; ++v)
          af[mb].u[v] = *(const unsigned int*)&Atile[arow + koffs[v]];
      }

      #pragma unroll
      for (int j = 0; j < 4; ++j) {
        const int ncol = (nBase + j * 16 + (lane & 15)) * 32;
        Bf16Frag bf;
        #pragma unroll
        for (int v = 0; v < 8; ++v)
          bf.u[v] = *(const unsigned int*)&Bb[ncol + koffs[v]];
        #pragma unroll
        for (int mb = 0; mb < 2; ++mb)
          acc[mb][j] = __builtin_amdgcn_wmma_f32_16x16x32_bf16(
              false, af[mb].v, false, bf.v, (short)0, acc[mb][j], false, false);
      }
      __syncthreads();                 // all reads of B[i&1] done before reuse
    }
  }

  // ---- epilogue: BN2 + ReLU + residual ----
  const int mlo = lane & 15;
  const int mhi = (lane >> 4) << 3;
  #pragma unroll
  for (int j = 0; j < 4; ++j) {
    const int co = nBase + j * 16 + mlo;
    const float sc = g2[co] * rsqrtf(v2[co] + BN_EPS);
    const float sh = b2[co] - m2[co] * sc;
    #pragma unroll
    for (int mb = 0; mb < 2; ++mb) {
      #pragma unroll
      for (int r = 0; r < 8; ++r) {
        const int row = wm * 32 + mb * 16 + r + mhi;
        const size_t idx = (size_t)(pixBase + row) * CC + co;
        float x = acc[mb][j][r] * sc + sh;
        x = fmaxf(x, 0.f) + fmapD[idx];
        out[idx] = x;
      }
    }
  }
}

// ============================================================
extern "C" void kernel_launch(void* const* d_in, const int* in_sizes, int n_in,
                              void* d_out, int out_size, void* d_ws, size_t ws_size,
                              hipStream_t stream) {
  (void)in_sizes; (void)n_in; (void)out_size; (void)ws_size;
  const float* fmapD = (const float*)d_in[0];
  const float* fmapS = (const float*)d_in[1];
  const float* w1 = (const float*)d_in[2];
  const float* g1 = (const float*)d_in[3];
  const float* b1 = (const float*)d_in[4];
  const float* m1 = (const float*)d_in[5];
  const float* v1 = (const float*)d_in[6];
  const float* w2 = (const float*)d_in[7];
  const float* g2 = (const float*)d_in[8];
  const float* b2 = (const float*)d_in[9];
  const float* m2 = (const float*)d_in[10];
  const float* v2 = (const float*)d_in[11];
  float* out = (float*)d_out;

  char* ws = (char*)d_ws;
  float* gate = (float*)ws;                                   // 32768*9*4 = 1,179,648 B
  unsigned short* w2p = (unsigned short*)(ws + 1179648);      // 589824*2 = 1,179,648 B

  gate_kernel<<<NPIX / 4, 256, 0, stream>>>(fmapS, w1, g1, b1, m1, v1, gate);
  w2prep_kernel<<<589824 / 256, 256, 0, stream>>>(w2, w2p);
  conv_kernel<<<NPIX / 64, 256, 0, stream>>>(fmapD, gate, w2p, g2, b2, m2, v2, out);
}
